// TTTAdapter_2070174236922
// MI455X (gfx1250) — compile-verified, matrix operand order checked by reading, not run on hardware
//
#include <hip/hip_runtime.h>
#include <hip/hip_bf16.h>
#include <math.h>

// ---------------- problem constants ----------------
#define Bc   8
#define Lc   4096
#define Dc   1024
#define Hc   2
#define Fc   128          // head dim f
#define Kc   16           // minibatch tokens
#define dC   256          // H*f
#define Mtok (Bc * Lc)    // 32768 tokens
#define NMb  (Lc / Kc)    // 256 minibatches

// ---------------- WMMA plumbing (CDNA5 gfx1250, wave32) ----------------
typedef __attribute__((ext_vector_type(16))) __bf16 v16bf;
typedef __attribute__((ext_vector_type(8)))  __bf16 v8bf;
typedef __attribute__((ext_vector_type(4)))  __bf16 v4bf;
typedef __attribute__((ext_vector_type(8)))  float  v8f;

static __device__ __forceinline__ v8f wmma_bf16(v16bf a, v16bf b, v8f c) {
  return __builtin_amdgcn_wmma_f32_16x16x32_bf16(false, a, false, b, (short)0, c, false, false);
}

static __device__ __forceinline__ v16bf cat16(v8bf lo, v8bf hi) {
  return __builtin_shufflevector(lo, hi, 0,1,2,3,4,5,6,7,8,9,10,11,12,13,14,15);
}

#define WREDUCE(s) { for (int o_ = 16; o_; o_ >>= 1) (s) += __shfl_xor((s), o_, 32); }

static __device__ __forceinline__ float fast_sigmoid(float v) {
  return __builtin_amdgcn_rcpf(1.0f + __expf(-v));   // v_rcp_f32, no IEEE div chain
}

// CDNA5 async global->LDS DMA (ASYNCcnt). GVS mode: saddr(SGPR64) + vaddr(VGPR u32 bytes)
static __device__ __forceinline__ void async_b128(unsigned lds_off, unsigned gbyte_off,
                                                  const void* base) {
  asm volatile("global_load_async_to_lds_b128 %0, %1, %2"
               :: "v"(lds_off), "v"(gbyte_off), "s"(base) : "memory");
}
static __device__ __forceinline__ void async_b64(unsigned lds_off, unsigned gbyte_off,
                                                 const void* base) {
  asm volatile("global_load_async_to_lds_b64 %0, %1, %2"
               :: "v"(lds_off), "v"(gbyte_off), "s"(base) : "memory");
}
static __device__ __forceinline__ void wait_async0() {
  asm volatile("s_wait_asynccnt 0" ::: "memory");
}
static __device__ __forceinline__ void wait_async3() {   // keep 1 prefetch chunk in flight
  asm volatile("s_wait_asynccnt 3" ::: "memory");
}
static __device__ __forceinline__ unsigned lds_off_of(const void* p) {
  return (unsigned)(size_t)p;   // generic LDS addr keeps the wg-relative offset in [31:0]
}

// ---------------- helper: f32 -> bf16 bulk convert ----------------
__global__ void __launch_bounds__(256) cvt_bf16_kernel(const float* __restrict__ X,
                                                       __bf16* __restrict__ Y) {
  size_t i = ((size_t)blockIdx.x * 256 + threadIdx.x) * 4;
  float4 v = *(const float4*)(X + i);
  v4bf o;
  o[0] = (__bf16)v.x; o[1] = (__bf16)v.y; o[2] = (__bf16)v.z; o[3] = (__bf16)v.w;
  *(v4bf*)(Y + i) = o;
}

// ---------------- helper: weight [K][N] f32 -> [N][K] bf16 (one-time) ----------------
__global__ void __launch_bounds__(256) wtrans_kernel(const float* __restrict__ W,
                                                     __bf16* __restrict__ Wt,
                                                     int K, int N) {
  int e = blockIdx.x * 256 + threadIdx.x;          // over N*K, coalesced writes
  int n = e / K, k = e - n * K;
  Wt[e] = (__bf16)W[(size_t)k * N + n];
}

// ---------------- GEMM: C = act(A[MxK]bf16 @ Wt[NxK]bf16 + bias) ----------------
// block = 256 threads (8 waves). Block tile 32x128, K-chunk 32.
// Double-buffered LDS; tiles staged raw via async global->LDS DMA (exactly 3 async
// instructions per wave per chunk: 1x b64 for A, 2x b128 for W), one chunk prefetched
// ahead so the DMA overlaps WMMA. Fragments = ds_load_b128 pairs, zero converts.
#define TSTR 56   // LDS row stride in bf16 elems: 112B = 16B-aligned, 28-bank step
__global__ void __launch_bounds__(256) gemm_bf16_kernel(
    const __bf16* __restrict__ A, const __bf16* __restrict__ Wt,
    const float* __restrict__ bias, float* __restrict__ C, __bf16* __restrict__ Cb,
    int M, int N, int Kd, int act)
{
  __shared__ __bf16 Asb[2 * 32 * TSTR];
  __shared__ __bf16 Wsb[2 * 128 * TSTR];
  const int tid  = threadIdx.x;
  const int lane = tid & 31;
  const int wave = tid >> 5;
  const int wm   = (wave >> 2) * 16;   // 0 or 16
  const int wn   = (wave & 3) * 32;    // 0,32,64,96
  const int m0   = blockIdx.x * 32;
  const int n0   = blockIdx.y * 128;
  const int frow = lane & 15;
  const int koff = (lane >> 4) * 8;
  const int crow = (lane >> 4) * 8;
  const int ccol = lane & 15;

  const unsigned asb0 = lds_off_of(Asb);
  const unsigned wsb0 = lds_off_of(Wsb);
  const int arow = tid >> 3, aseg = tid & 7;       // A tile: 32 rows x 8x8B segs
  const int wrow0 = tid >> 2, wseg = tid & 3;      // W tile: 128 rows x 4x16B segs

  auto stage = [&](int kcc, int sel) {
    unsigned ab = asb0 + (unsigned)sel * (32 * TSTR * 2);
    unsigned wb = wsb0 + (unsigned)sel * (128 * TSTR * 2);
    async_b64(ab + arow * (TSTR * 2) + aseg * 8,
              (unsigned)(((size_t)(m0 + arow) * Kd + kcc) * 2) + aseg * 8, A);
    async_b128(wb + wrow0 * (TSTR * 2) + wseg * 16,
               (unsigned)(((size_t)(n0 + wrow0) * Kd + kcc) * 2) + wseg * 16, Wt);
    async_b128(wb + (wrow0 + 64) * (TSTR * 2) + wseg * 16,
               (unsigned)(((size_t)(n0 + wrow0 + 64) * Kd + kcc) * 2) + wseg * 16, Wt);
  };

  v8f acc0 = {}; v8f acc1 = {};
  stage(0, 0);                                      // prologue prefetch
  for (int kc = 0; kc < Kd; kc += 32) {
    const int cur = (kc >> 5) & 1;
    if (kc + 32 < Kd) { stage(kc + 32, cur ^ 1); wait_async3(); }
    else              { wait_async0(); }
    __syncthreads();

    const __bf16* ar = &Asb[cur * 32 * TSTR + (wm + frow) * TSTR];
    const __bf16* b0 = &Wsb[cur * 128 * TSTR + (wn + ccol) * TSTR];
    const __bf16* b1 = &Wsb[cur * 128 * TSTR + (wn + 16 + ccol) * TSTR];
    v16bf af  = cat16(*(const v8bf*)(ar + koff), *(const v8bf*)(ar + 16 + koff));
    v16bf bf0 = cat16(*(const v8bf*)(b0 + koff), *(const v8bf*)(b0 + 16 + koff));
    v16bf bf1 = cat16(*(const v8bf*)(b1 + koff), *(const v8bf*)(b1 + 16 + koff));
    acc0 = wmma_bf16(af, bf0, acc0);
    acc1 = wmma_bf16(af, bf1, acc1);
    __syncthreads();
  }
#pragma unroll
  for (int r = 0; r < 8; ++r) {
    int gm  = m0 + wm + crow + r;
    int gn0 = n0 + wn + ccol;
    int gn1 = gn0 + 16;
    float v0 = acc0[r];
    float v1 = acc1[r];
    if (bias) { v0 += bias[gn0]; v1 += bias[gn1]; }
    if (act)  { v0 *= fast_sigmoid(v0); v1 *= fast_sigmoid(v1); }
    if (C)  { C[(size_t)gm * N + gn0] = v0;          C[(size_t)gm * N + gn1] = v1; }
    if (Cb) { Cb[(size_t)gm * N + gn0] = (__bf16)v0; Cb[(size_t)gm * N + gn1] = (__bf16)v1; }
  }
}

// ---------------- row LayerNorm over 256 cols; 1 wave / row; f32 and/or bf16 out ----------
__global__ void __launch_bounds__(256) ln_kernel(
    const float* __restrict__ X, const float* __restrict__ g,
    const float* __restrict__ b, float* __restrict__ Y, __bf16* __restrict__ Yb, float eps)
{
  const int lane = threadIdx.x & 31;
  const int wave = threadIdx.x >> 5;
  const size_t r = (size_t)blockIdx.x * 8 + wave;
  const float* xr = X + r * dC;
  float v[8]; float s = 0.f;
#pragma unroll
  for (int j = 0; j < 8; ++j) { v[j] = xr[lane + j * 32]; s += v[j]; }
  WREDUCE(s);
  float mu = s * (1.0f / dC);
  float q = 0.f;
#pragma unroll
  for (int j = 0; j < 8; ++j) { float t = v[j] - mu; q += t * t; }
  WREDUCE(q);
  float inv = rsqrtf(q * (1.0f / dC) + eps);
#pragma unroll
  for (int j = 0; j < 8; ++j) {
    int c = lane + j * 32;
    float o = g[c] * (v[j] - mu) * inv + b[c];
    if (Y)  Y[r * dC + c] = o;
    if (Yb) Yb[r * dC + c] = (__bf16)o;
  }
}

// ---------------- per-token TTT learning rates (reads bf16 h) ----------------
__global__ void __launch_bounds__(256) lrv_kernel(
    const __bf16* __restrict__ Hb, const float* __restrict__ lr_w,
    const float* __restrict__ lr_b, float* __restrict__ lrv)
{
  const int lane = threadIdx.x & 31;
  const int wave = threadIdx.x >> 5;
  const size_t t = (size_t)blockIdx.x * 8 + wave;
  const __bf16* hr = Hb + t * dC;
#pragma unroll
  for (int h = 0; h < Hc; ++h) {
    float s = 0.f;
#pragma unroll
    for (int j = 0; j < 8; ++j) { int c = lane + j * 32; s += (float)hr[c] * lr_w[h * dC + c]; }
    WREDUCE(s);
    if (lane == 0)
      lrv[(size_t)h * Mtok + t] = fast_sigmoid(s + lr_b[h]) * (1.0f / (float)Fc);
  }
}

// ---------------- sequential TTT scan: 1 workgroup per (b,h) chain ----------------
__global__ void __launch_bounds__(256) scan_kernel(
    const float* __restrict__ XQ, const float* __restrict__ XK,
    const float* __restrict__ XV, const float* __restrict__ lrv,
    const float* __restrict__ W1g, const float* __restrict__ b1g,
    const float* __restrict__ lnw, const float* __restrict__ lnb,
    const float* __restrict__ tok_idx, float* __restrict__ Out)
{
  extern __shared__ float smem[];
  float* W1s  = smem;                    // 128 x 132
  float* xqs  = W1s  + 128 * 132;        // 16 x 132
  float* xks  = xqs  + 16 * 132;
  float* xvs  = xks  + 16 * 132;
  float* Z1s  = xvs  + 16 * 132;
  float* grs  = Z1s  + 16 * 132;
  float* zbs  = grs  + 16 * 132;
  float* Attn = zbs  + 16 * 132;         // 16 x 16
  float* M2s  = Attn + 256;              // 16 x 16 (pre-negated)
  float* b1s  = M2s  + 256;              // 128
  float* gws  = b1s  + 128;              // 128
  float* gbs  = gws  + 128;              // 128
  float* lrvs = gbs  + 128;              // 16
  float* tkvs = lrvs + 16;               // 16

  const int tid  = threadIdx.x;
  const int lane = tid & 31;
  const int wave = tid >> 5;
  const int bb   = blockIdx.x >> 1;      // batch
  const int hh   = blockIdx.x & 1;       // head
  const int frow = lane & 15;
  const int koff = (lane >> 4) * 8;
  const int crow = (lane >> 4) * 8;
  const int ccol = lane & 15;

  for (int e = tid; e < Fc * Fc; e += 256)
    W1s[(e >> 7) * 132 + (e & 127)] = W1g[(size_t)hh * Fc * Fc + e];
  if (tid < Fc) {
    b1s[tid] = b1g[hh * Fc + tid];
    gws[tid] = lnw[hh * Fc + tid];
    gbs[tid] = lnb[hh * Fc + tid];
  }
  __syncthreads();

  for (int n = 0; n < NMb; ++n) {
    const size_t t0 = (size_t)bb * Lc + (size_t)n * Kc;
#pragma unroll
    for (int j = 0; j < 8; ++j) {
      int e = tid + j * 256;
      int r = e >> 7, c = e & 127;
      size_t gi = (t0 + r) * dC + (size_t)hh * Fc + c;
      xqs[r * 132 + c] = XQ[gi];
      xks[r * 132 + c] = XK[gi];
      xvs[r * 132 + c] = XV[gi];
    }
    if (tid < Kc) {
      lrvs[tid] = lrv[(size_t)hh * Mtok + t0 + tid];
      float ti = 1.0f / (float)(tid + 1) + tok_idx[tid];
      tkvs[tid] = ti > 0.f ? ti : 0.f;
    }
    __syncthreads();

    // Phase 1: Z1 = xk @ W1 + b1
    {
      v8f acc = {};
#pragma unroll
      for (int kc = 0; kc < 4; ++kc) {
        v16bf af, bfr;
#pragma unroll
        for (int i = 0; i < 8; ++i) {
          af[i]    = (__bf16)xks[frow * 132 + kc * 32 + koff + i];
          af[8+i]  = (__bf16)xks[frow * 132 + kc * 32 + 16 + koff + i];
          bfr[i]   = (__bf16)W1s[(kc * 32 + koff + i) * 132 + wave * 16 + ccol];
          bfr[8+i] = (__bf16)W1s[(kc * 32 + 16 + koff + i) * 132 + wave * 16 + ccol];
        }
        acc = wmma_bf16(af, bfr, acc);
      }
#pragma unroll
      for (int r = 0; r < 8; ++r)
        Z1s[(crow + r) * 132 + wave * 16 + ccol] = acc[r] + b1s[wave * 16 + ccol];
    }
    // Attn = tril(xq @ xk^T) (wave 0)
    if (wave == 0) {
      v8f acc = {};
#pragma unroll
      for (int kc = 0; kc < 4; ++kc) {
        v16bf af, bfr;
#pragma unroll
        for (int i = 0; i < 8; ++i) {
          af[i]    = (__bf16)xqs[frow * 132 + kc * 32 + koff + i];
          af[8+i]  = (__bf16)xqs[frow * 132 + kc * 32 + 16 + koff + i];
          bfr[i]   = (__bf16)xks[ccol * 132 + kc * 32 + koff + i];
          bfr[8+i] = (__bf16)xks[ccol * 132 + kc * 32 + 16 + koff + i];
        }
        acc = wmma_bf16(af, bfr, acc);
      }
#pragma unroll
      for (int r = 0; r < 8; ++r) {
        int i = crow + r, j = ccol;
        Attn[i * 16 + j] = (j <= i) ? acc[r] : 0.0f;
      }
    }
    __syncthreads();

    // Phase 2: grad = LN-fused-L2 backward; 2 rows per wave
#pragma unroll
    for (int rr = 0; rr < 2; ++rr) {
      int i = wave * 2 + rr;
      float x[4], tg[4];
      float s = 0.f;
#pragma unroll
      for (int j = 0; j < 4; ++j) {
        int c = lane * 4 + j;
        x[j]  = Z1s[i * 132 + c];
        tg[j] = xvs[i * 132 + c] - xks[i * 132 + c];
        s += x[j];
      }
      WREDUCE(s);
      float mu = s * (1.0f / Fc);
      float q = 0.f;
#pragma unroll
      for (int j = 0; j < 4; ++j) { float t = x[j] - mu; q += t * t; }
      WREDUCE(q);
      float std  = sqrtf(q * (1.0f / Fc) + 1e-6f);
      float rstd = 1.0f / std;
      float gy[4]; float sg = 0.f, sgx = 0.f;
#pragma unroll
      for (int j = 0; j < 4; ++j) {
        int c = lane * 4 + j;
        float xh = (x[j] - mu) * rstd;
        float g  = gws[c];
        float v  = (g * xh + gbs[c] - tg[j]) * g;
        gy[j] = v; sg += v; sgx += v * xh;
      }
      WREDUCE(sg); WREDUCE(sgx);
#pragma unroll
      for (int j = 0; j < 4; ++j) {
        int c = lane * 4 + j;
        float xh = (x[j] - mu) * rstd;
        grs[i * 132 + c] = ((float)Fc * gy[j] - sg - xh * sgx) * (rstd * (1.0f / Fc));
      }
    }
    {
      int i = tid >> 4, j = tid & 15;
      float m2 = (j <= i) ? tkvs[i] * lrvs[j] * (Attn[i * 16 + j] + 1.0f) : 0.0f;
      M2s[i * 16 + j] = -m2;
    }
    __syncthreads();

    // Phase 3: Z1_bar = xq@W1 - M2@grad + b1
    {
      v8f acc = {};
#pragma unroll
      for (int kc = 0; kc < 4; ++kc) {
        v16bf af, bfr;
#pragma unroll
        for (int i = 0; i < 8; ++i) {
          af[i]    = (__bf16)xqs[frow * 132 + kc * 32 + koff + i];
          af[8+i]  = (__bf16)xqs[frow * 132 + kc * 32 + 16 + koff + i];
          bfr[i]   = (__bf16)W1s[(kc * 32 + koff + i) * 132 + wave * 16 + ccol];
          bfr[8+i] = (__bf16)W1s[(kc * 32 + 16 + koff + i) * 132 + wave * 16 + ccol];
        }
        acc = wmma_bf16(af, bfr, acc);
      }
      v16bf af2, bf2;   // K padded 16 -> 32 with zeros
#pragma unroll
      for (int i = 0; i < 8; ++i) {
        int k1 = koff + i;
        af2[i]   = (__bf16)M2s[frow * 16 + k1];
        af2[8+i] = (__bf16)0.0f;
        bf2[i]   = (__bf16)grs[k1 * 132 + wave * 16 + ccol];
        bf2[8+i] = (__bf16)0.0f;
      }
      acc = wmma_bf16(af2, bf2, acc);
#pragma unroll
      for (int r = 0; r < 8; ++r)
        zbs[(crow + r) * 132 + wave * 16 + ccol] = acc[r] + b1s[wave * 16 + ccol];
    }
    __syncthreads();

    // Phase 4: out = xq + LN(Z1_bar)
#pragma unroll
    for (int rr = 0; rr < 2; ++rr) {
      int i = wave * 2 + rr;
      float z[4]; float s = 0.f;
#pragma unroll
      for (int j = 0; j < 4; ++j) { z[j] = zbs[i * 132 + lane * 4 + j]; s += z[j]; }
      WREDUCE(s);
      float mu = s * (1.0f / Fc);
      float q = 0.f;
#pragma unroll
      for (int j = 0; j < 4; ++j) { float t = z[j] - mu; q += t * t; }
      WREDUCE(q);
      float rstd = rsqrtf(q * (1.0f / Fc) + 1e-6f);
#pragma unroll
      for (int j = 0; j < 4; ++j) {
        int c = lane * 4 + j;
        Out[(t0 + i) * dC + (size_t)hh * Fc + c] =
            xqs[i * 132 + c] + gws[c] * (z[j] - mu) * rstd + gbs[c];
      }
    }

    // Phase 5: W1 -= (last_eta*xk)^T @ grad ; b1 -= sum
    {
      const float le = tkvs[Kc - 1];
#pragma unroll
      for (int tn = 0; tn < 8; ++tn) {
        v8f acc;
#pragma unroll
        for (int r = 0; r < 8; ++r)
          acc[r] = W1s[(wave * 16 + crow + r) * 132 + tn * 16 + ccol];
        v16bf af, bfr;
#pragma unroll
        for (int i = 0; i < 8; ++i) {
          int tok = koff + i;
          af[i]    = (__bf16)(-le * lrvs[tok] * xks[tok * 132 + wave * 16 + frow]);
          af[8+i]  = (__bf16)0.0f;
          bfr[i]   = (__bf16)grs[tok * 132 + tn * 16 + ccol];
          bfr[8+i] = (__bf16)0.0f;
        }
        acc = wmma_bf16(af, bfr, acc);
#pragma unroll
        for (int r = 0; r < 8; ++r)
          W1s[(wave * 16 + crow + r) * 132 + tn * 16 + ccol] = acc[r];
      }
      if (tid < Fc) {
        float s = 0.f;
#pragma unroll
        for (int i = 0; i < Kc; ++i) s += le * lrvs[i] * grs[i * 132 + tid];
        b1s[tid] -= s;
      }
    }
    __syncthreads();
  }
}

// ---------------- host-side orchestration ----------------
extern "C" void kernel_launch(void* const* d_in, const int* in_sizes, int n_in,
                              void* d_out, int out_size, void* d_ws, size_t ws_size,
                              hipStream_t stream) {
  const float* x      = (const float*)d_in[0];
  const float* down_w = (const float*)d_in[1];
  const float* down_b = (const float*)d_in[2];
  const float* up_w   = (const float*)d_in[3];
  const float* up_b   = (const float*)d_in[4];
  const float* wq     = (const float*)d_in[5];
  const float* wk     = (const float*)d_in[6];
  const float* wv     = (const float*)d_in[7];
  const float* wo     = (const float*)d_in[8];
  const float* W1     = (const float*)d_in[9];
  const float* b1     = (const float*)d_in[10];
  const float* tlnw   = (const float*)d_in[11];
  const float* tlnb   = (const float*)d_in[12];
  const float* lr_w   = (const float*)d_in[13];
  const float* lr_b   = (const float*)d_in[14];
  const float* tok    = (const float*)d_in[15];
  const float* postw  = (const float*)d_in[16];
  const float* postb  = (const float*)d_in[17];
  const float* lnw    = (const float*)d_in[18];
  const float* lnb    = (const float*)d_in[19];
  float* out = (float*)d_out;

  // ---- workspace carve-up (bytes) ----
  char* w = (char*)d_ws;
  __bf16* xb     = (__bf16*)w;                 w += (size_t)Mtok * Dc * 2;   // 67 MB
  __bf16* hbufb  = (__bf16*)w;                 w += (size_t)Mtok * dC * 2;   // 16 MB
  float*  qbuf   = (float*)w;                  w += (size_t)Mtok * dC * 4;   // 33 MB
  float*  kbuf   = (float*)w;                  w += (size_t)Mtok * dC * 4;
  float*  vbuf   = (float*)w;                  w += (size_t)Mtok * dC * 4;
  float*  obuf   = (float*)w;                  w += (size_t)Mtok * dC * 4;
  float*  lrvb   = (float*)w;                  w += (size_t)Hc * Mtok * 4;
  __bf16* lnpostb= (__bf16*)w;                 w += (size_t)Mtok * dC * 2;
  __bf16* hlnb   = (__bf16*)w;                 w += (size_t)Mtok * dC * 2;
  __bf16* dwt    = (__bf16*)w;                 w += (size_t)Dc * dC * 2;     // [256][1024]
  __bf16* qt     = (__bf16*)w;                 w += (size_t)dC * dC * 2;
  __bf16* kt     = (__bf16*)w;                 w += (size_t)dC * dC * 2;
  __bf16* vt     = (__bf16*)w;                 w += (size_t)dC * dC * 2;
  __bf16* wot    = (__bf16*)w;                 w += (size_t)dC * dC * 2;
  __bf16* upt    = (__bf16*)w;                 w += (size_t)dC * Dc * 2;     // [1024][256]
  float*  tttout = qbuf;                       // reuse (dead after scan)

  // ---- one-time operand preparation (bf16 everywhere) ----
  cvt_bf16_kernel<<<(Mtok * Dc) / (256 * 4), 256, 0, stream>>>(x, xb);
  wtrans_kernel<<<(Dc * dC) / 256, 256, 0, stream>>>(down_w, dwt, Dc, dC);
  wtrans_kernel<<<(dC * dC) / 256, 256, 0, stream>>>(wq, qt, dC, dC);
  wtrans_kernel<<<(dC * dC) / 256, 256, 0, stream>>>(wk, kt, dC, dC);
  wtrans_kernel<<<(dC * dC) / 256, 256, 0, stream>>>(wv, vt, dC, dC);
  wtrans_kernel<<<(dC * dC) / 256, 256, 0, stream>>>(wo, wot, dC, dC);
  wtrans_kernel<<<(dC * Dc) / 256, 256, 0, stream>>>(up_w, upt, dC, Dc);

  // ---- 1. down-proj + silu -> bf16 h ----
  gemm_bf16_kernel<<<dim3(Mtok / 32, dC / 128), 256, 0, stream>>>(
      xb, dwt, down_b, nullptr, hbufb, Mtok, dC, Dc, /*silu*/1);

  // ---- 2. QKV projections (f32 out for the scan) ----
  gemm_bf16_kernel<<<dim3(Mtok / 32, dC / 128), 256, 0, stream>>>(
      hbufb, qt, nullptr, qbuf, nullptr, Mtok, dC, dC, 0);
  gemm_bf16_kernel<<<dim3(Mtok / 32, dC / 128), 256, 0, stream>>>(
      hbufb, kt, nullptr, kbuf, nullptr, Mtok, dC, dC, 0);
  gemm_bf16_kernel<<<dim3(Mtok / 32, dC / 128), 256, 0, stream>>>(
      hbufb, vt, nullptr, vbuf, nullptr, Mtok, dC, dC, 0);

  // ---- 3. per-token learning rates ----
  lrv_kernel<<<Mtok / 8, 256, 0, stream>>>(hbufb, lr_w, lr_b, lrvb);

  // ---- 4. sequential TTT scan (W1 state in LDS, ~122 KB/WGP) ----
  constexpr int SMEM_SCAN = (128 * 132 + 6 * (16 * 132) + 256 + 256 + 3 * 128 + 32) * 4;
  (void)hipFuncSetAttribute((const void*)scan_kernel,
                            hipFuncAttributeMaxDynamicSharedMemorySize, SMEM_SCAN);
  scan_kernel<<<Bc * Hc, 256, SMEM_SCAN, stream>>>(
      qbuf, kbuf, vbuf, lrvb, W1, b1, tlnw, tlnb, tok, obuf);

  // ---- 5. post-LN -> wo GEMM -> LN -> up GEMM ----
  ln_kernel<<<Mtok / 8, 256, 0, stream>>>(obuf, postw, postb, nullptr, lnpostb, 1e-5f);
  gemm_bf16_kernel<<<dim3(Mtok / 32, dC / 128), 256, 0, stream>>>(
      lnpostb, wot, nullptr, tttout, nullptr, Mtok, dC, dC, 0);
  ln_kernel<<<Mtok / 8, 256, 0, stream>>>(tttout, lnw, lnb, nullptr, hlnb, 1e-5f);
  gemm_bf16_kernel<<<dim3(Mtok / 32, Dc / 128), 256, 0, stream>>>(
      hlnb, upt, up_b, out, nullptr, Mtok, Dc, dC, 0);
}
